// reservoir_computing_80075370266953
// MI455X (gfx1250) — matrix operand order repaired and checked by reading, not compiled, not run on hardware
//
#include <hip/hip_runtime.h>
#include <hip/hip_bf16.h>

#define NV     207
#define DR     512
#define LSTEPS 128
#define BATCH  16
#define RHO_C   1.0f
#define DELTA_C 0.1f
#define BBIAS_C 0.0f

#define NKC1   16          // K chunks GEMM1 over e (512/32)
#define NKC2   7           // K chunks GEMM2 over v (224/32, NV padded)
#define NNT    13          // v/u tiles (208/16, NV padded)
#define NGT    32          // d tiles over DR (512/16)
#define TLD    232         // LDS row stride (halves) for T^T [d][v]
#define SROW   40          // staged state row stride in halves (64B data + 16B pad)

typedef __attribute__((ext_vector_type(16))) _Float16 v16h;
typedef __attribute__((ext_vector_type(8)))  _Float16 v8h;
typedef __attribute__((ext_vector_type(8)))  float    v8f;
typedef __attribute__((ext_vector_type(4)))  unsigned int u32x4;
typedef __attribute__((ext_vector_type(4)))  int          i32x4;
typedef __attribute__((ext_vector_type(8)))  int          i32x8;

#if defined(__has_builtin)
#  if __has_builtin(__builtin_amdgcn_tensor_load_to_lds)
#    define HAVE_TDM 1
#  endif
#endif
#ifndef HAVE_TDM
#  define HAVE_TDM 0
#endif

#if HAVE_TDM
// Issue one TDM transfer: 2D f16 tile (32 cols x 207 rows, tensor row stride
// 512), LDS padding 16B after every 64B row -> 80B LDS row stride.
__device__ __forceinline__ void tdm_load_tile(unsigned lds_addr, const _Float16* gptr)
{
    const unsigned long long ga = (unsigned long long)(uintptr_t)gptr;
    u32x4 g0;
    g0[0] = 1u;                                              // count=1 (user D#)
    g0[1] = lds_addr;                                        // LDS byte address
    g0[2] = (unsigned)(ga & 0xFFFFFFFFu);                    // global_addr lo
    g0[3] = (unsigned)((ga >> 32) & 0x01FFFFFFu) | (2u << 30); // ga hi | type=2
    i32x8 g1;
    g1[0] = (1 << 16)        // data_size = 1 -> 2 bytes
          | (1 << 20)        // pad_enable
          | (3 << 22)        // pad_interval: 16 DWORDs (64B)
          | (3 << 25);       // pad_amount:   4 DWORDs (16B)
    g1[1] = (DR << 16);      // tensor_dim0[15:0] in bits[63:48]
    g1[2] = (NV << 16);      // tensor_dim1[15:0] in bits[95:80]
    g1[3] = (32 << 16);      // tile_dim0 = 32   in bits[127:112]
    g1[4] = NV;              // tile_dim1 = 207  in bits[143:128]
    g1[5] = DR;              // tensor_dim0_stride lo32 in bits[191:160]
    g1[6] = 0;
    g1[7] = 0;
    i32x4 gz; gz[0] = gz[1] = gz[2] = gz[3] = 0;
#  if __has_include(<hip/amd_detail/amd_gfx1250_TDM.h>)
    i32x8 gz8; gz8[0]=gz8[1]=gz8[2]=gz8[3]=gz8[4]=gz8[5]=gz8[6]=gz8[7]=0;
    __builtin_amdgcn_tensor_load_to_lds(g0, g1, gz, gz, gz8, 0);   // clang-23
#  else
    __builtin_amdgcn_tensor_load_to_lds(g0, g1, gz, gz, 0);        // ROCm 7.2
#  endif
}
#endif

// ---------------------------------------------------------------------------
// Prep 1: R -> f16 WMMA-A fragments of R^T (R^T[d,e] = R[e*DR + d]).
// ---------------------------------------------------------------------------
__global__ __launch_bounds__(256)
void prep_rtfrag(const float* __restrict__ R, _Float16* __restrict__ dst)
{
    const int idx = blockIdx.x * 256 + threadIdx.x;
    if (idx >= NGT * NKC1 * 32) return;
    const int lane = idx & 31, frag = idx >> 5;
    const int kc = frag & (NKC1 - 1), g = frag >> 4;
    const int lh = lane & 15, hi = lane >> 4;
    const int d  = g * 16 + lh;
    const int k0 = kc * 32 + hi * 16;
    v16h v;
#pragma unroll
    for (int q = 0; q < 8; ++q) {
        v[q]     = (_Float16)R[(long)(k0 + q) * DR + d];
        v[q + 8] = (_Float16)R[(long)(k0 + 8 + q) * DR + d];
    }
    ((v16h*)dst)[idx] = v;
}

// ---------------------------------------------------------------------------
// Prep 2: IN -> f16 WMMA-B fragments (GEMM2), NV-pad zeros baked in.
// ---------------------------------------------------------------------------
__global__ __launch_bounds__(256)
void prep_infrag(const float* __restrict__ IN, _Float16* __restrict__ dst)
{
    const int idx = blockIdx.x * 256 + threadIdx.x;
    if (idx >= NNT * NKC2 * 32) return;
    const int lane = idx & 31, frag = idx >> 5;
    const int kc = frag % NKC2, nt = frag / NKC2;
    const int lh = lane & 15, hi = lane >> 4;
    const int u  = nt * 16 + lh;
    const int k0 = kc * 32 + hi * 16;
    v16h x;
#pragma unroll
    for (int j = 0; j < 16; ++j) {
        const int v = k0 + j;
        x[j] = (u < NV && v < NV) ? (_Float16)IN[(long)v * NV + u] : (_Float16)0.f;
    }
    ((v16h*)dst)[idx] = x;
}

// ---------------------------------------------------------------------------
// Prep 3: initial state f32 -> f16 row-major copy.
// ---------------------------------------------------------------------------
__global__ __launch_bounds__(256)
void prep_state16(const float* __restrict__ src, _Float16* __restrict__ dst, int n8)
{
    const int idx = blockIdx.x * 256 + threadIdx.x;
    if (idx >= n8) return;
    const float* s = src + (long)idx * 8;
    v8h o;
#pragma unroll
    for (int q = 0; q < 8; ++q) o[q] = (_Float16)s[q];
    ((v8h*)dst)[idx] = o;
}

// ---------------------------------------------------------------------------
// One step (both GEMMs transposed):
//   T^T = R^T @ S^T     (M=d 32/WG, K=e 512, N=v 208) -> LDS [d][v] f16
//   rec^T = T^T @ IN    (M=d 32/WG, K=v 224, N=u 208)
//   out = tanh(rec + DELTA * x[u] * win[d]); also write next f16 state.
// S^T k-chunks are streamed into LDS by the TDM (double-buffered).
// ---------------------------------------------------------------------------
__global__ __launch_bounds__(256)
void reservoir_step(const float*    __restrict__ test_data,
                    const _Float16* __restrict__ rtfrag,
                    const _Float16* __restrict__ infrag,
                    const float*    __restrict__ win,
                    float*          __restrict__ out,
                    const _Float16* __restrict__ sIn,
                    _Float16*       __restrict__ sOut,
                    int t)
{
    __shared__ _Float16 Ttr[32 * TLD];          // T^T: [d 0..31][v 0..223]
#if HAVE_TDM
    __shared__ _Float16 stage[2][NV * SROW];    // staged S^T chunks (80B rows)
#endif

    const int nblk = blockIdx.x;
    const int b    = blockIdx.y;
    const int tid  = threadIdx.x;
    const int lane = tid & 31;
    const int lh   = lane & 15;
    const int hi   = lane >> 4;
    // wave-uniform ids as scalars: keeps tile guards off the EXEC mask
    const int wv   = __builtin_amdgcn_readfirstlane(tid >> 5);   // 0..7
    const int m    = wv & 1;               // d-tile within slab
    const int g    = nblk * 2 + m;         // global d-tile
    const int ngrp = wv >> 1;              // v/u tiles: ngrp, +4, +8, +12
    const int dl   = m * 16 + lh;

    const _Float16* sb = sIn + (long)b * NV * DR;

    // zero T^T K-pad columns v = 208..223 (must be finite for WMMA)
    for (int i = tid; i < 32 * 16; i += 256)
        Ttr[(i >> 4) * TLD + 208 + (i & 15)] = (_Float16)0.f;

    const int nt[4] = { ngrp, ngrp + 4, ngrp + 8, ngrp + 12 };
    const int nnt   = (ngrp == 0) ? 4 : 3;          // scalar

    v8f zero8;
#pragma unroll
    for (int i = 0; i < 8; ++i) zero8[i] = 0.0f;

    // ---------------- GEMM1: T^T = R^T @ S^T ----------------
    v8f acc[4];
#pragma unroll
    for (int j = 0; j < 4; ++j) acc[j] = zero8;

    const v16h* rf = (const v16h*)rtfrag + ((long)g * NKC1) * 32 + lane;

#if HAVE_TDM
    const unsigned lds0 = (unsigned)(uintptr_t)&stage[0][0];
    const unsigned lds1 = (unsigned)(uintptr_t)&stage[1][0];
    if (wv == 0) {                               // one TDM op per k-chunk
        tdm_load_tile(lds0, sb);
        tdm_load_tile(lds1, sb + 32);
    }
    for (int kc = 0; kc < NKC1; ++kc) {
        if (wv == 0) {
            if (kc == NKC1 - 1) __builtin_amdgcn_s_wait_tensorcnt(0);
            else                __builtin_amdgcn_s_wait_tensorcnt(1);
        }
        __syncthreads();                         // chunk kc is in LDS
        const v16h arf = rf[kc * 32];
        const _Float16* sbuf = &stage[kc & 1][0];
#pragma unroll
        for (int j = 0; j < 4; ++j) {
            if (j < nnt) {
                int v = nt[j] * 16 + lh;
                v = (v < NV) ? v : (NV - 1);
                const _Float16* bp = sbuf + v * SROW + hi * 16;
                const v8h blo = *(const v8h*)bp;
                const v8h bhi = *(const v8h*)(bp + 8);
                v16h bf;
#pragma unroll
                for (int q = 0; q < 8; ++q) { bf[q] = blo[q]; bf[q + 8] = bhi[q]; }
                acc[j] = __builtin_amdgcn_wmma_f32_16x16x32_f16(
                    false, arf, false, bf, (short)0, acc[j], false, false);
            }
        }
        __syncthreads();                         // everyone done with buffer
        if (wv == 0 && kc + 2 < NKC1)
            tdm_load_tile((kc & 1) ? lds1 : lds0, sb + (long)(kc + 2) * 32);
    }
#else
    for (int kc = 0; kc < NKC1; ++kc) {
        const v16h arf = rf[kc * 32];
        const int  eb  = kc * 32 + hi * 16;
#pragma unroll
        for (int j = 0; j < 4; ++j) {
            if (j < nnt) {
                int v = nt[j] * 16 + lh;
                v = (v < NV) ? v : (NV - 1);
                const v16h bf = *(const v16h*)(sb + (long)v * DR + eb);
                acc[j] = __builtin_amdgcn_wmma_f32_16x16x32_f16(
                    false, arf, false, bf, (short)0, acc[j], false, false);
            }
        }
    }
#endif

    // scatter T^T into LDS [d][v]
#pragma unroll
    for (int j = 0; j < 4; ++j) {
        if (j < nnt) {
            const int v = nt[j] * 16 + lh;
#pragma unroll
            for (int i = 0; i < 8; ++i)
                Ttr[(m * 16 + hi * 8 + i) * TLD + v] = (_Float16)(acc[j][i] * RHO_C);
        }
    }
    __syncthreads();

    // ---------------- GEMM2: rec^T = T^T @ IN ----------------
    v8f acc2[4];
#pragma unroll
    for (int j = 0; j < 4; ++j) acc2[j] = zero8;

    for (int kc = 0; kc < NKC2; ++kc) {
        const int k0 = kc * 32 + hi * 16;
        const v8h alo = *(const v8h*)&Ttr[dl * TLD + k0];
        const v8h ahi = *(const v8h*)&Ttr[dl * TLD + k0 + 8];
        v16h af;
#pragma unroll
        for (int q = 0; q < 8; ++q) { af[q] = alo[q]; af[q + 8] = ahi[q]; }

#pragma unroll
        for (int j = 0; j < 4; ++j) {
            if (j < nnt) {
                const v16h bf = ((const v16h*)infrag)[((long)nt[j] * NKC2 + kc) * 32 + lane];
                acc2[j] = __builtin_amdgcn_wmma_f32_16x16x32_f16(
                    false, af, false, bf, (short)0, acc2[j], false, false);
            }
        }
    }

    // ------------- epilogue: tanh(rec + DELTA*x[u]*win[d]) -------------
    const int   dbase = g * 16 + hi * 8;
    const v8f   w8    = *(const v8f*)(win + dbase);
    const float* xrow = test_data + ((long)b * LSTEPS + t) * (long)NV;
    float*       outp = out + ((long)b * LSTEPS + t) * (long)NV * DR;
    _Float16*    sop  = sOut + (long)b * NV * DR;

#pragma unroll
    for (int j = 0; j < 4; ++j) {
        if (j < nnt) {
            const int u  = nt[j] * 16 + lh;
            const float xd = xrow[(u < NV) ? u : (NV - 1)] * DELTA_C;
            v8f o8;
            v8h h8;
#pragma unroll
            for (int i = 0; i < 8; ++i) {
                const float y = tanhf(acc2[j][i] + xd * w8[i] + BBIAS_C);
                o8[i] = y;
                h8[i] = (_Float16)y;
            }
            if (u < NV) {
                *(v8f*)(outp + (long)u * DR + dbase) = o8;
                *(v8h*)(sop  + (long)u * DR + dbase) = h8;
            }
        }
    }
}

extern "C" void kernel_launch(void* const* d_in, const int* in_sizes, int n_in,
                              void* d_out, int out_size, void* d_ws, size_t ws_size,
                              hipStream_t stream) {
    (void)in_sizes; (void)n_in; (void)ws_size; (void)out_size;
    const float* test_data = (const float*)d_in[0];
    const float* Rnet      = (const float*)d_in[1];
    const float* INnet     = (const float*)d_in[2];
    const float* win       = (const float*)d_in[3];
    const float* state0    = (const float*)d_in[4];
    float*       out       = (float*)d_out;

    const long RT_H = (long)NGT * NKC1 * 32 * 16;
    const long IN_H = (long)NNT * NKC2 * 32 * 16;
    const long ST_H = (long)BATCH * NV * DR;

    _Float16* rtfrag = (_Float16*)d_ws;
    _Float16* infrag = rtfrag + RT_H;
    _Float16* st0    = infrag + IN_H;
    _Float16* st1    = st0 + ST_H;

    {
        const int n1 = NGT * NKC1 * 32;
        prep_rtfrag<<<(n1 + 255) / 256, 256, 0, stream>>>(Rnet, rtfrag);
        const int n2 = NNT * NKC2 * 32;
        prep_infrag<<<(n2 + 255) / 256, 256, 0, stream>>>(INnet, infrag);
        const int n3 = (int)(ST_H / 8);
        prep_state16<<<(n3 + 255) / 256, 256, 0, stream>>>(state0, st0, n3);
    }

    const dim3 grid(DR / 32, BATCH);
    const dim3 block(256);

    for (int t = 0; t < LSTEPS; ++t) {
        const _Float16* sIn  = (t & 1) ? st1 : st0;
        _Float16*       sOut = (t & 1) ? st0 : st1;
        reservoir_step<<<grid, block, 0, stream>>>(
            test_data, rtfrag, infrag, win, out, sIn, sOut, t);
    }
}
